// PrototypeSimilarity_17428977287839
// MI455X (gfx1250) — compile-verified, matrix operand order checked by reading, not run on hardware
//
#include <hip/hip_runtime.h>
#include <math.h>

// ---------------------------------------------------------------------------
// Cosine similarity: out(N,C) = <x_n, w_c> / (max(|x_n|,eps) * max(|w_c|,eps))
// N=32768, D=512, C=2048 (f32 in, f32 out).
//
// Strategy (MI455X / gfx1250, wave32):
//  - Kernel 1: per-row inverse norms (1/max(||row||,1e-8)) for X and W -> ws.
//  - Kernel 2: 128x128 output tile per 256-thread block; K-loop staged through
//    LDS as BF16; V_WMMA_F32_16X16X32_BF16 accumulation in F32; epilogue
//    scales accumulators by invX[m]*invW[c].
// ---------------------------------------------------------------------------

typedef __attribute__((ext_vector_type(16))) __bf16 v16bf;
typedef __attribute__((ext_vector_type(8)))  __bf16 v8bf;
typedef __attribute__((ext_vector_type(4)))  __bf16 v4bf;
typedef __attribute__((ext_vector_type(8)))  float  v8f;

#define D_DIM 512
#define BM 128
#define BN 128
#define BK 64
#define LDS_STRIDE 72   // 64 bf16 + 8 pad -> 144B row stride (conflict-free)

// ---------------- Kernel 1: inverse row norms ------------------------------
__global__ __launch_bounds__(256)
void inv_norms_kernel(const float* __restrict__ X, const float* __restrict__ W,
                      float* __restrict__ invX, float* __restrict__ invW,
                      int N, int C)
{
    const int wave = (int)((blockIdx.x * blockDim.x + threadIdx.x) >> 5);
    const int lane = (int)(threadIdx.x & 31);
    const int total = N + C;
    if (wave >= total) return;

    const float* row = (wave < N) ? (X + (size_t)wave * D_DIM)
                                  : (W + (size_t)(wave - N) * D_DIM);
    float s = 0.0f;
#pragma unroll
    for (int j = 0; j < 4; ++j) {
        // lane l reads float4 at element 4*(l + 32*j): 512B contiguous per j
        float4 v = *(const float4*)(row + 4 * (lane + 32 * j));
        s += v.x * v.x + v.y * v.y + v.z * v.z + v.w * v.w;
    }
#pragma unroll
    for (int off = 16; off > 0; off >>= 1)
        s += __shfl_xor(s, off, 32);

    if (lane == 0) {
        float inv = 1.0f / fmaxf(sqrtf(s), 1e-8f);
        if (wave < N) invX[wave] = inv;
        else          invW[wave - N] = inv;
    }
}

// ---------------- Kernel 2: WMMA GEMM + scale epilogue ---------------------
__global__ __launch_bounds__(256)
void cosine_wmma_kernel(const float* __restrict__ X, const float* __restrict__ W,
                        const float* __restrict__ invX, const float* __restrict__ invW,
                        float* __restrict__ Out, int C)
{
    __shared__ __align__(16) __bf16 lA[BM * LDS_STRIDE];
    __shared__ __align__(16) __bf16 lB[BN * LDS_STRIDE];

    const int m0   = (int)blockIdx.y * BM;
    const int n0   = (int)blockIdx.x * BN;
    const int tid  = (int)threadIdx.x;
    const int wave = tid >> 5;
    const int lane = tid & 31;
    const int wm   = wave & 3;        // wave row  (4): 32 M-rows each
    const int wn   = wave >> 2;       // wave col  (2): 64 N-cols each
    const int lhalf = lane >> 4;      // 0: K {0..7,16..23}, 1: K {8..15,24..31}
    const int l16   = lane & 15;

    v8f acc[2][4];
#pragma unroll
    for (int i = 0; i < 2; ++i)
#pragma unroll
        for (int j = 0; j < 4; ++j)
            acc[i][j] = (v8f){};

    for (int k0 = 0; k0 < D_DIM; k0 += BK) {
        // ---- stage 128x64 f32 of X and W -> bf16 LDS tiles (256 threads) ----
#pragma unroll
        for (int i = 0; i < 8; ++i) {
            const int idx = tid + i * 256;    // 0..2047
            const int row = idx >> 4;         // 0..127
            const int c4  = idx & 15;         // float4 column within 64 K
            float4 va = *(const float4*)(X + (size_t)(m0 + row) * D_DIM + k0 + 4 * c4);
            float4 vb = *(const float4*)(W + (size_t)(n0 + row) * D_DIM + k0 + 4 * c4);
            v4bf pa = { (__bf16)va.x, (__bf16)va.y, (__bf16)va.z, (__bf16)va.w };
            v4bf pb = { (__bf16)vb.x, (__bf16)vb.y, (__bf16)vb.z, (__bf16)vb.w };
            *(v4bf*)(lA + row * LDS_STRIDE + 4 * c4) = pa;   // ds_store_b64
            *(v4bf*)(lB + row * LDS_STRIDE + 4 * c4) = pb;
        }
        __syncthreads();

        // ---- two K=32 WMMA steps over the staged BK=64 ----
#pragma unroll
        for (int kk = 0; kk < BK; kk += 32) {
            v16bf afrag[2];
            v16bf bfrag[4];
#pragma unroll
            for (int t = 0; t < 2; ++t) {
                const __bf16* p = lA + (wm * 32 + t * 16 + l16) * LDS_STRIDE
                                     + kk + lhalf * 8;
                v8bf lo = *(const v8bf*)(p);        // K+0..7  (or 8..15)
                v8bf hi = *(const v8bf*)(p + 16);   // K+16..23 (or 24..31)
                afrag[t] = __builtin_shufflevector(lo, hi,
                            0,1,2,3,4,5,6,7,8,9,10,11,12,13,14,15);
            }
#pragma unroll
            for (int t = 0; t < 4; ++t) {
                const __bf16* p = lB + (wn * 64 + t * 16 + l16) * LDS_STRIDE
                                     + kk + lhalf * 8;
                v8bf lo = *(const v8bf*)(p);
                v8bf hi = *(const v8bf*)(p + 16);
                bfrag[t] = __builtin_shufflevector(lo, hi,
                            0,1,2,3,4,5,6,7,8,9,10,11,12,13,14,15);
            }
#pragma unroll
            for (int i = 0; i < 2; ++i)
#pragma unroll
                for (int j = 0; j < 4; ++j)
                    acc[i][j] = __builtin_amdgcn_wmma_f32_16x16x32_bf16(
                        /*neg_a=*/false, afrag[i],
                        /*neg_b=*/false, bfrag[j],
                        /*c_mod=*/(short)0, acc[i][j],
                        /*reuse_a=*/false, /*reuse_b=*/false);
        }
        __syncthreads();
    }

    // ---- epilogue: out[m][c] = acc * invX[m] * invW[c] ----
    // C/D layout: VGPR r, lane l -> M = r + 8*(l>=16), N = l%16.
#pragma unroll
    for (int i = 0; i < 2; ++i) {
        const int rbase = m0 + wm * 32 + i * 16 + lhalf * 8;  // 8-aligned
        float4 s0 = *(const float4*)(invX + rbase);
        float4 s1 = *(const float4*)(invX + rbase + 4);
        float sx[8] = { s0.x, s0.y, s0.z, s0.w, s1.x, s1.y, s1.z, s1.w };
#pragma unroll
        for (int j = 0; j < 4; ++j) {
            const int col = n0 + wn * 64 + j * 16 + l16;
            const float sw = invW[col];
#pragma unroll
            for (int r = 0; r < 8; ++r) {
                Out[(size_t)(rbase + r) * (size_t)C + col] = acc[i][j][r] * sx[r] * sw;
            }
        }
    }
}

// ---------------------------------------------------------------------------
extern "C" void kernel_launch(void* const* d_in, const int* in_sizes, int n_in,
                              void* d_out, int out_size, void* d_ws, size_t ws_size,
                              hipStream_t stream) {
    const float* X = (const float*)d_in[0];        // (N, 512)
    const float* W = (const float*)d_in[1];        // (C, 512)
    float* out = (float*)d_out;                    // (N, C)

    const int N = in_sizes[0] / D_DIM;             // 32768
    const int C = in_sizes[1] / D_DIM;             // 2048

    float* invX = (float*)d_ws;                    // N floats
    float* invW = invX + N;                        // C floats  (~136 KB total)

    const int totalRows = N + C;                   // one wave per row
    const int nb = (totalRows + 7) / 8;            // 8 waves per 256-thr block
    hipLaunchKernelGGL(inv_norms_kernel, dim3(nb), dim3(256), 0, stream,
                       X, W, invX, invW, N, C);

    dim3 grid(C / BN, N / BM);                     // (16, 256)
    hipLaunchKernelGGL(cosine_wmma_kernel, grid, dim3(256), 0, stream,
                       X, W, invX, invW, out, C);
}
